// KANLayer_14328010900290
// MI455X (gfx1250) — matrix operand order repaired and checked by reading, not compiled, not run on hardware
//
#include <hip/hip_runtime.h>
#include <stddef.h>

// ---------------------------------------------------------------------------
// KAN layer for MI455X (gfx1250): 13-term bf16 GEMM (silu + Chebyshev T1..T12)
// via v_wmma_f32_16x16x32_bf16 with fp32 accumulation; the T0==1 term is a
// constant bias (column sums of pw[0]) added in the epilogue; RMSNorm finish.
// ---------------------------------------------------------------------------

typedef __attribute__((ext_vector_type(16))) __bf16 v16bf;
typedef __attribute__((ext_vector_type(8)))  float  v8f;

#define IN_F   1024
#define OUT_F  1024
#define N_TERM 14          // base + 13 Chebyshev orders (ord 1 == T0 hoisted)
#define BM     128
#define BN     128
#define KCH    32          // x-columns per K step (bf16 WMMA K=32)
#define LDA    40          // LDS row stride in bf16 (32 + 8 pad, 80B = 16B-mult)

__device__ __forceinline__ __bf16 to_bf16(float f) { return (__bf16)f; }

// ---------------------------------------------------------------------------
// Kernel 1: repack fp32 weights into bf16, pre-tiled in WMMA B-fragment order.
// Fragment tile = (ord, kc, nt): 32 lanes x 16 bf16, contiguous 1 KB.
//   lane = ((k%32)/16)*16 + (n%16),  j = k%16  ->  element j of lane's v16bf
// ---------------------------------------------------------------------------
__global__ __launch_bounds__(256)
void kan_repack_weights(const float* __restrict__ base_w,
                        const float* __restrict__ poly_w,
                        __bf16* __restrict__ wt) {
    size_t idx = (size_t)blockIdx.x * blockDim.x + threadIdx.x;
    if (idx >= (size_t)N_TERM * IN_F * OUT_F) return;
    const int n   = (int)(idx & (OUT_F - 1));
    const int k   = (int)((idx >> 10) & (IN_F - 1));
    const int ord = (int)(idx >> 20);

    float v = (ord == 0)
        ? base_w[(size_t)k * OUT_F + n]
        : poly_w[((size_t)(ord - 1) * IN_F + k) * OUT_F + n];

    const int kc   = k >> 5;
    const int kin  = k & 31;
    const int j    = kin & 15;
    const int lane = ((kin >> 4) << 4) | (n & 15);
    const int nt   = n >> 4;
    const size_t dst = ((((size_t)ord * 32 + kc) * (OUT_F / 16) + nt) * 32 + lane) * 16 + j;
    wt[dst] = to_bf16(v);
}

// ---------------------------------------------------------------------------
// Kernel 1b: bias[n] = sum_k pw[0][k][n]   (the T0 == ones term, hoisted)
// ---------------------------------------------------------------------------
__global__ __launch_bounds__(256)
void kan_bias(const float* __restrict__ poly_w, float* __restrict__ bias) {
    const int n = blockIdx.x * blockDim.x + threadIdx.x;
    if (n >= OUT_F) return;
    float s = 0.0f;
    for (int k = 0; k < IN_F; ++k)
        s += poly_w[(size_t)k * OUT_F + n];         // pw[0] (fp32 weights
    bias[n] = (float)to_bf16(s);                    //  summed; keep bf16 rounding
}

// ---------------------------------------------------------------------------
// Kernel 2: fused activation + GEMM. Grid: (OUT_F/BN, rows/BM), 256 threads.
// 8 waves (2 along M x 4 along N), wave tile 64x32 -> acc[4][2] of v8f.
// B fragments reused 4x from registers; A fragments from double-buffered LDS.
// ---------------------------------------------------------------------------
__global__ __launch_bounds__(256)
void kan_gemm(const float* __restrict__ x,     // [rows, IN_F] fp32
              const __bf16* __restrict__ wt,   // repacked fragments
              const float* __restrict__ bias,  // [OUT_F]
              float* __restrict__ out) {       // [rows, OUT_F] fp32 (pre-norm)
    __shared__ __bf16 a_tile[2][BM * LDA];

    const int tid    = threadIdx.x;
    const int lane   = tid & 31;
    const int wave   = tid >> 5;
    const int wm     = wave & 1;        // wave row 0..1 (64 rows each)
    const int wn     = wave >> 1;       // wave col 0..3 (32 cols each)
    const int blk_n  = blockIdx.x;      // 0..OUT_F/BN-1
    const int blk_m  = blockIdx.y;      // 0..rows/BM-1

    // activation staging ownership: thread owns 16 consecutive x elements
    const int own_row = tid >> 1;                 // 0..127
    const int own_ch  = (tid & 1) * 16;           // 0 or 16 within the 32-chunk

    // A-fragment addressing (16-bit A 16x32 layout)
    const int sel  = lane >> 4;                   // 0: K {0..15}, 1: {16..31}
    const int arow = lane & 15;

    v8f acc[4][2];
    #pragma unroll
    for (int mi = 0; mi < 4; ++mi)
        #pragma unroll
        for (int ni = 0; ni < 2; ++ni)
            #pragma unroll
            for (int r = 0; r < 8; ++r)
                acc[mi][ni][r] = 0.0f;

    const float* xrow = x + (size_t)(blk_m * BM + own_row) * IN_F + own_ch;

    for (int kc = 0; kc < IN_F / KCH; ++kc) {
        // ---- load 16 owned x values, compute tanh once ----
        float xv[16];
        #pragma unroll
        for (int i = 0; i < 4; ++i) {
            const float4 v = *(const float4*)(xrow + kc * KCH + i * 4);
            xv[i * 4 + 0] = v.x; xv[i * 4 + 1] = v.y;
            xv[i * 4 + 2] = v.z; xv[i * 4 + 3] = v.w;
        }
        float th[16];
        #pragma unroll
        for (int i = 0; i < 16; ++i) th[i] = tanhf(xv[i]);

        float Tp[16], Tc[16];

        for (int ord = 0; ord < N_TERM; ++ord) {
            // T0 == ones: rank-degenerate GEMM term hoisted into bias epilogue.
            // Uniform across the block -> barrier pairing preserved.
            if (ord == 1) {
                #pragma unroll
                for (int i = 0; i < 16; ++i) Tp[i] = 1.0f;
                continue;
            }
            const int p = ord & 1;

            // ---- produce 16 bf16 A values for this term ----
            union { __bf16 h[16]; uint4 u[2]; } s;
            if (ord == 0) {                      // silu(x) @ base_weight
                #pragma unroll
                for (int i = 0; i < 16; ++i)
                    s.h[i] = to_bf16(xv[i] / (1.0f + __expf(-xv[i])));
            } else if (ord == 2) {               // T1 = tanh(x)
                #pragma unroll
                for (int i = 0; i < 16; ++i) { Tc[i] = th[i]; s.h[i] = to_bf16(th[i]); }
            } else {                             // T_k = 2 t T_{k-1} - T_{k-2}
                #pragma unroll
                for (int i = 0; i < 16; ++i) {
                    const float tn = 2.0f * th[i] * Tc[i] - Tp[i];
                    Tp[i] = Tc[i]; Tc[i] = tn;
                    s.h[i] = to_bf16(tn);
                }
            }

            // ---- stage A tile (double buffered: ONE barrier per term) ----
            __bf16* dst = &a_tile[p][own_row * LDA + own_ch];
            *(uint4*)(dst)     = s.u[0];
            *(uint4*)(dst + 8) = s.u[1];
            __syncthreads();

            // ---- load A fragments from LDS ----
            union { v16bf v; uint4 u[2]; } af[4];
            #pragma unroll
            for (int mi = 0; mi < 4; ++mi) {
                const __bf16* ab = &a_tile[p][(wm * 64 + mi * 16 + arow) * LDA];
                af[mi].u[0] = *(const uint4*)(ab + sel * 16);       // K base0
                af[mi].u[1] = *(const uint4*)(ab + sel * 16 + 8);   // K base0+8
            }

            // prefetch the next used term's B tile into L2 (global_prefetch_b8)
            if (ord + 1 < N_TERM) {
                const int nxt = (ord == 0) ? 2 : ord + 1;
                const int nt0 = blk_n * (BN / 16) + wn * 2;
                __builtin_prefetch(
                    wt + ((((size_t)nxt * 32 + kc) * (OUT_F / 16) + nt0) * 32 + lane) * 16,
                    0, 1);
            }

            // ---- B fragments (pre-tiled, coalesced) + 8 WMMA per wave ----
            #pragma unroll
            for (int ni = 0; ni < 2; ++ni) {
                const int nt = blk_n * (BN / 16) + wn * 2 + ni;
                const __bf16* bp =
                    wt + ((((size_t)ord * 32 + kc) * (OUT_F / 16) + nt) * 32 + lane) * 16;
                union { v16bf v; uint4 u[2]; } bf;
                bf.u[0] = *(const uint4*)(bp);
                bf.u[1] = *(const uint4*)(bp + 8);
                #pragma unroll
                for (int mi = 0; mi < 4; ++mi) {
                    acc[mi][ni] = __builtin_amdgcn_wmma_f32_16x16x32_bf16(
                        false, af[mi].v, false, bf.v,
                        (short)0, acc[mi][ni], false, false);
                }
            }
            // no trailing barrier: double buffering + next term's store->read
            // barrier orders ord's reads before ord+2's writes, except ord 0 -> 2 (ord 1 skips its barrier).
        }
    }

    // ---- epilogue: add hoisted T0 bias, store fp32 pre-norm tile ----
    // C layout: VGPR r -> M = r + 8*sel (lane group), N = lane & 15
    #pragma unroll
    for (int mi = 0; mi < 4; ++mi) {
        #pragma unroll
        for (int ni = 0; ni < 2; ++ni) {
            const int col = blk_n * BN + wn * 32 + ni * 16 + (lane & 15);
            const float b = bias[col];
            #pragma unroll
            for (int r = 0; r < 8; ++r) {
                const int row = blk_m * BM + wm * 64 + mi * 16 + sel * 8 + r;
                out[(size_t)row * OUT_F + col] = acc[mi][ni][r] + b;
            }
        }
    }
}

// ---------------------------------------------------------------------------
// Kernel 3: in-place RMSNorm over each row of out (wave32 shuffle reduction).
// ---------------------------------------------------------------------------
__global__ __launch_bounds__(256)
void kan_rmsnorm(float* __restrict__ out, const float* __restrict__ scale) {
    const int row  = blockIdx.x;
    const int tid  = threadIdx.x;
    const int lane = tid & 31;
    const int wave = tid >> 5;
    float* p = out + (size_t)row * OUT_F;

    float v[4];
    float ss = 0.0f;
    #pragma unroll
    for (int i = 0; i < 4; ++i) {
        v[i] = p[tid + i * 256];
        ss += v[i] * v[i];
    }
    #pragma unroll
    for (int off = 16; off > 0; off >>= 1)
        ss += __shfl_xor(ss, off, 32);

    __shared__ float wsum[8];
    if (lane == 0) wsum[wave] = ss;
    __syncthreads();
    float tot = 0.0f;
    #pragma unroll
    for (int w = 0; w < 8; ++w) tot += wsum[w];

    const float r = rsqrtf(tot * (1.0f / OUT_F) + 1e-6f);
    #pragma unroll
    for (int i = 0; i < 4; ++i)
        p[tid + i * 256] = v[i] * r * scale[tid + i * 256];
}

// ---------------------------------------------------------------------------
extern "C" void kernel_launch(void* const* d_in, const int* in_sizes, int n_in,
                              void* d_out, int out_size, void* d_ws, size_t ws_size,
                              hipStream_t stream) {
    const float* x      = (const float*)d_in[0];
    const float* base_w = (const float*)d_in[1];
    const float* poly_w = (const float*)d_in[2];
    const float* scale  = (const float*)d_in[3];
    float* out          = (float*)d_out;

    __bf16* wt  = (__bf16*)d_ws;                          // 28 MiB bf16 fragments
    float* bias = (float*)((char*)d_ws +
                           (size_t)N_TERM * IN_F * OUT_F * sizeof(__bf16));

    const int rows = in_sizes[0] / IN_F;                  // 8192

    // 1) repack fp32 weights -> bf16 WMMA B-fragment layout in workspace
    {
        const size_t total = (size_t)N_TERM * IN_F * OUT_F;
        const int blocks = (int)((total + 255) / 256);
        kan_repack_weights<<<blocks, 256, 0, stream>>>(base_w, poly_w, wt);
    }
    // 1b) hoisted T0 bias: column sums of pw[0]
    kan_bias<<<(OUT_F + 255) / 256, 256, 0, stream>>>(poly_w, bias);

    // 2) fused activation + 13-term GEMM -> pre-norm fp32 in d_out
    {
        dim3 grid(OUT_F / BN, rows / BM);
        kan_gemm<<<grid, 256, 0, stream>>>(x, wt, bias, out);
    }
    // 3) in-place RMSNorm
    kan_rmsnorm<<<rows, 256, 0, stream>>>(out, scale);
}
